// Encoder_50242527428945
// MI455X (gfx1250) — matrix operand order repaired and checked by reading, not compiled, not run on hardware
//
#include <hip/hip_runtime.h>
#include <hip/hip_bf16.h>

#define N_NODES 50000
#define N_EDGES 800000
#define C       128
#define ROWS    64          // rows per block in the GEMM kernel
#define BATCH   1024

typedef _Float16 f16;
typedef __attribute__((ext_vector_type(16))) _Float16 v16h;
typedef __attribute__((ext_vector_type(8)))  _Float16 v8h;
typedef __attribute__((ext_vector_type(8)))  float    v8f;

// ---------------------------------------------------------------------------
// Degree: one f32 atomic per edge (computed once, reused by all 3 layers).
// ---------------------------------------------------------------------------
__global__ void deg_kernel(const long long* __restrict__ dst,
                           float* __restrict__ deg) {
    int e = blockIdx.x * blockDim.x + threadIdx.x;
    if (e < N_EDGES) atomicAdd(&deg[dst[e]], 1.0f);
}

// ---------------------------------------------------------------------------
// Scatter-add aggregation: one wave (32 lanes) per edge, lane handles 4
// channels -> coalesced float4 gather of x[src] (L2-resident: x is 25.6MB
// inside the 192MB L2), 4 f32 atomics into msg[dst].
// ---------------------------------------------------------------------------
__global__ void scatter_kernel(const float* __restrict__ x,
                               const long long* __restrict__ src,
                               const long long* __restrict__ dst,
                               float* __restrict__ msg) {
    int gtid = blockIdx.x * blockDim.x + threadIdx.x;
    int e    = gtid >> 5;            // wave id == edge id
    int lane = gtid & 31;
    if (e >= N_EDGES) return;
    long long s = src[e];
    long long d = dst[e];
    float4 v = ((const float4*)(x + (size_t)s * C))[lane];
    float* md = msg + (size_t)d * C + 4 * lane;
    atomicAdd(md + 0, v.x);
    atomicAdd(md + 1, v.y);
    atomicAdd(md + 2, v.z);
    atomicAdd(md + 3, v.w);
}

// ---------------------------------------------------------------------------
// Fused SAGE layer: out = PReLU( (msg/deg) @ Wl^T + bl + x @ Wr^T )
// 256 threads = 8 waves. Block: 64 rows x 128 cols. Wave w -> cols [16w,16w+16).
// All GEMM math on v_wmma_f32_16x16x32_f16 (f16 in, f32 accumulate).
// ---------------------------------------------------------------------------
__global__ __launch_bounds__(256)
void sage_layer_kernel(const float* __restrict__ msg,
                       const float* __restrict__ deg,
                       const float* __restrict__ xin,
                       const float* __restrict__ Wl,
                       const float* __restrict__ bl,
                       const float* __restrict__ Wr,
                       const float* __restrict__ slope,
                       float* __restrict__ xout) {
    __shared__ alignas(32) f16 sWl[C * C];      // Wl[c][k] as f16   (32 KB)
    __shared__ alignas(32) f16 sWr[C * C];      // Wr[c][k] as f16   (32 KB)
    __shared__ alignas(32) f16 sAgg[ROWS * C];  // (msg/deg) tile    (16 KB)
    __shared__ alignas(32) f16 sAx[ROWS * C];   // x tile            (16 KB)

    const int tid  = threadIdx.x;
    const int row0 = blockIdx.x * ROWS;

    // Stage weights (reused by every K-step / M-tile in this block).
    for (int i = tid; i < C * C; i += 256) {
        sWl[i] = (f16)Wl[i];
        sWr[i] = (f16)Wr[i];
    }
    // Stage A tiles; fuse the mean-divide into the f32->f16 convert.
    for (int i = tid; i < ROWS * C; i += 256) {
        int r  = i >> 7;
        int k  = i & (C - 1);
        int gr = row0 + r;
        float av = 0.0f, xv = 0.0f;
        if (gr < N_NODES) {
            float d = deg[gr];
            d  = d > 1.0f ? d : 1.0f;
            av = msg[(size_t)gr * C + k] / d;
            xv = xin[(size_t)gr * C + k];
        }
        sAgg[i] = (f16)av;
        sAx[i]  = (f16)xv;
    }
    __syncthreads();

    const int wave = tid >> 5;
    const int lane = tid & 31;
    const int n    = lane & 15;        // tile column
    const int hi   = lane >> 4;        // lane half (0: lanes 0-15, 1: 16-31)
    const int col0 = wave * 16;
    const int c    = col0 + n;

    const f16* wlRow = &sWl[c * C];
    const f16* wrRow = &sWr[c * C];
    const float bias = bl[c];
    const float sl   = slope[c];

    #pragma unroll
    for (int mt = 0; mt < ROWS / 16; ++mt) {
        const f16* aggRow = &sAgg[(mt * 16 + n) * C];
        const f16* axRow  = &sAx [(mt * 16 + n) * C];
        v8f acc = {};
        #pragma unroll
        for (int ks = 0; ks < 4; ++ks) {        // K = 128 in 4 steps of 32
            const int kb   = ks * 32;
            // A fragment (contiguous K, matching B): lanes 0-15 hold
            // K in [0..15]; lanes 16-31 hold K in [16..31].
            const int aoff = kb + (hi ? 16 : 0);
            v8h aLo = *(const v8h*)(aggRow + aoff);
            v8h aHi = *(const v8h*)(aggRow + aoff + 8);
            v8h xLo = *(const v8h*)(axRow  + aoff);
            v8h xHi = *(const v8h*)(axRow  + aoff + 8);
            v16h aAgg, aX;
            #pragma unroll
            for (int j = 0; j < 8; ++j) {
                aAgg[j] = aLo[j]; aAgg[8 + j] = aHi[j];
                aX[j]   = xLo[j]; aX[8 + j]   = xHi[j];
            }
            // B fragment: lane holds 16 contiguous K for column n,
            // lanes 0-15 -> K[0..15], lanes 16-31 -> K[16..31]. B[k][c]=W[c][k],
            // so it is a contiguous 32-byte run of the (f16) weight row.
            const int boff = kb + (hi ? 16 : 0);
            v16h bL = *(const v16h*)(wlRow + boff);
            v16h bR = *(const v16h*)(wrRow + boff);
            acc = __builtin_amdgcn_wmma_f32_16x16x32_f16(
                      false, aAgg, false, bL, (short)0, acc, false, false);
            acc = __builtin_amdgcn_wmma_f32_16x16x32_f16(
                      false, aX,   false, bR, (short)0, acc, false, false);
        }
        // C/D layout: VGPR v, lane -> row = v + 8*hi, col = lane%16.
        #pragma unroll
        for (int v = 0; v < 8; ++v) {
            int gr = row0 + mt * 16 + v + (hi ? 8 : 0);
            if (gr < N_NODES) {
                float val = acc[v] + bias;
                val = val > 0.0f ? val : sl * val;
                xout[(size_t)gr * C + c] = val;
            }
        }
    }
}

__global__ void copy_out_kernel(const float* __restrict__ in,
                                float* __restrict__ out, int n) {
    int i = blockIdx.x * blockDim.x + threadIdx.x;
    if (i < n) out[i] = in[i];
}

// ---------------------------------------------------------------------------
extern "C" void kernel_launch(void* const* d_in, const int* in_sizes, int n_in,
                              void* d_out, int out_size, void* d_ws, size_t ws_size,
                              hipStream_t stream) {
    const float*     x  = (const float*)d_in[0];
    const long long* ei = (const long long*)d_in[1];   // [2, N_EDGES] int64
    const long long* src = ei;
    const long long* dst = ei + N_EDGES;
    // d_in[2] = batch_size (1024, known statically)

    const float* Wl[3] = {(const float*)d_in[3],  (const float*)d_in[7],  (const float*)d_in[11]};
    const float* bl[3] = {(const float*)d_in[4],  (const float*)d_in[8],  (const float*)d_in[12]};
    const float* Wr[3] = {(const float*)d_in[5],  (const float*)d_in[9],  (const float*)d_in[13]};
    const float* sa[3] = {(const float*)d_in[6],  (const float*)d_in[10], (const float*)d_in[14]};

    // Workspace layout: msg | x0 | x1 | deg  (~77 MB)
    const size_t NC = (size_t)N_NODES * C;
    float* msg  = (float*)d_ws;
    float* xb0  = msg + NC;
    float* xb1  = xb0 + NC;
    float* deg  = xb1 + NC;

    // Degree (shared by all layers).
    hipMemsetAsync(deg, 0, N_NODES * sizeof(float), stream);
    deg_kernel<<<(N_EDGES + 255) / 256, 256, 0, stream>>>(dst, deg);

    const int scatterBlocks = (N_EDGES * 32 + 255) / 256;   // 1 wave per edge
    const int gemmBlocks    = (N_NODES + ROWS - 1) / ROWS;

    const float* cur = x;
    float* bufs[2] = {xb0, xb1};
    for (int l = 0; l < 3; ++l) {
        hipMemsetAsync(msg, 0, NC * sizeof(float), stream);
        scatter_kernel<<<scatterBlocks, 256, 0, stream>>>(cur, src, dst, msg);
        float* nxt = bufs[l & 1];
        sage_layer_kernel<<<gemmBlocks, 256, 0, stream>>>(
            msg, deg, cur, Wl[l], bl[l], Wr[l], sa[l], nxt);
        cur = nxt;
    }

    copy_out_kernel<<<(BATCH * C + 255) / 256, 256, 0, stream>>>(
        cur, (float*)d_out, BATCH * C);
}
